// LinearInt8_18726057411189
// MI455X (gfx1250) — matrix operand order repaired and checked by reading, not compiled, not run on hardware
//
#include <hip/hip_runtime.h>
#include <hip/hip_fp16.h>
#include <stdint.h>

// Problem constants (from reference)
#define K_DIM   4096
#define N_DIM   11008
#define NG      32            // K_DIM / 128 quant groups

// GEMM tiling
#define BM      128
#define BN      256
#define BK      64            // K-chunk staged per iteration
#define LDA     72            // halves per LDS row (64 + 8 pad -> 144B stride, conflict-free)
#define LDB     72
#define THREADS 256           // 8 waves; 2(M) x 4(N) wave grid of 64x64 wave tiles

typedef __attribute__((ext_vector_type(16))) _Float16 v16h;
typedef __attribute__((ext_vector_type(8)))  _Float16 v8h;
typedef __attribute__((ext_vector_type(4)))  _Float16 v4h;
typedef __attribute__((ext_vector_type(8)))  float    v8f;

// ============================================================================
// Phase A: one-time conversions into workspace (pure-bandwidth passes).
// ============================================================================

// fp32 activations -> fp16  (grid-stride over float4 chunks)
__global__ __launch_bounds__(256)
void convertA_kernel(const float* __restrict__ A, _Float16* __restrict__ Ah, int total4)
{
    int idx = blockIdx.x * blockDim.x + threadIdx.x;
    const int stride = gridDim.x * blockDim.x;
    for (; idx < total4; idx += stride) {
        const float4 v = ((const float4*)A)[idx];
        v4h h;
        h[0] = (_Float16)v.x; h[1] = (_Float16)v.y;
        h[2] = (_Float16)v.z; h[3] = (_Float16)v.w;
        *(v4h*)(Ah + (size_t)idx * 4) = h;
    }
}

// int8 weights * group scale -> fp16  (grid-stride over 16-byte chunks)
// W is [N][K]; 256 chunks per row; quant group = 128 elems = 8 chunks.
__global__ __launch_bounds__(256)
void dequantW_kernel(const int8_t* __restrict__ W, const float* __restrict__ S,
                     _Float16* __restrict__ Wh, int total16)
{
    int idx = blockIdx.x * blockDim.x + threadIdx.x;
    const int stride = gridDim.x * blockDim.x;
    for (; idx < total16; idx += stride) {
        const int4 raw = ((const int4*)W)[idx];
        const int8_t* b = (const int8_t*)&raw;
        const int n    = idx >> 8;          // / (K_DIM/16)
        const int c16  = idx & 255;         // chunk within row
        const float s  = S[(size_t)n * NG + (c16 >> 3)];
        v8h h0, h1;
        #pragma unroll
        for (int e = 0; e < 8; ++e) {
            h0[e] = (_Float16)(s * (float)b[e]);
            h1[e] = (_Float16)(s * (float)b[e + 8]);
        }
        _Float16* d = Wh + (size_t)idx * 16;
        *(v8h*)(d)     = h0;
        *(v8h*)(d + 8) = h1;
    }
}

// ============================================================================
// Phase B: fp16 x fp16 -> fp32 WMMA GEMM (pure-copy LDS staging, no VALU).
// Block tile 128M x 256N, 8 waves of 64x64.
// ============================================================================
__global__ __launch_bounds__(THREADS)
void gemm_f16_kernel(const _Float16* __restrict__ Ah,   // [M, K] fp16
                     const _Float16* __restrict__ Wh,   // [N, K] fp16 (dequantized)
                     float* __restrict__ C)             // [M, N] fp32
{
    __shared__ _Float16 ldsA[BM * LDA];   // 18.4 KB
    __shared__ _Float16 ldsB[BN * LDB];   // 36.9 KB

    const int tid    = threadIdx.x;
    const int wid    = tid >> 5;
    const int lane   = tid & 31;
    const int laneHi = lane >> 4;
    const int lane15 = lane & 15;

    const int mBlock = blockIdx.y * BM;
    const int nBlock = blockIdx.x * BN;

    const int waveM = (wid & 1) * 64;     // 2(M) x 4(N) wave grid
    const int waveN = (wid >> 1) * 64;

    v8f acc[4][4] = {};                   // 64x64 per wave (128 VGPRs)

    // Register staging: A 128x64 halves -> 1024 x 16B / 256 thr = 4 each
    //                   B 256x64 halves -> 2048 x 16B / 256 thr = 8 each
    int4 aReg[4], bReg[8];

    auto loadChunk = [&](int k0) {
        #pragma unroll
        for (int j = 0; j < 4; ++j) {
            const int idx = tid + j * THREADS;  // 0..1023
            const int row = idx >> 3;           // 8 x 16B per row
            const int ch  = (idx & 7) * 8;      // half-index within row
            aReg[j] = *(const int4*)(Ah + (size_t)(mBlock + row) * K_DIM + k0 + ch);
        }
        #pragma unroll
        for (int j = 0; j < 8; ++j) {
            const int idx = tid + j * THREADS;  // 0..2047
            const int row = idx >> 3;
            const int ch  = (idx & 7) * 8;
            bReg[j] = *(const int4*)(Wh + (size_t)(nBlock + row) * K_DIM + k0 + ch);
        }
    };

    auto storeChunk = [&]() {
        #pragma unroll
        for (int j = 0; j < 4; ++j) {
            const int idx = tid + j * THREADS;
            const int row = idx >> 3;
            const int ch  = (idx & 7) * 8;
            *(int4*)(ldsA + row * LDA + ch) = aReg[j];
        }
        #pragma unroll
        for (int j = 0; j < 8; ++j) {
            const int idx = tid + j * THREADS;
            const int row = idx >> 3;
            const int ch  = (idx & 7) * 8;
            *(int4*)(ldsB + row * LDB + ch) = bReg[j];
        }
    };

    const int numChunks = K_DIM / BK;     // 64
    loadChunk(0);

    for (int kc = 0; kc < numChunks; ++kc) {
        storeChunk();                     // waits on outstanding global loads
        __syncthreads();
        if (kc + 1 < numChunks) loadChunk((kc + 1) * BK);  // overlap with WMMAs

        #pragma unroll
        for (int kk = 0; kk < 2; ++kk) {
            const int kb = kk * 32;

            // A frags: lanes 0-15: M=lane, halves = K(kb+0..7 | kb+16..23)
            //          lanes16-31: M=lane-16, halves = K(kb+8..15 | kb+24..31)
            v16h afrag[4];
            #pragma unroll
            for (int mi = 0; mi < 4; ++mi) {
                const _Float16* p = ldsA + (waveM + mi * 16 + lane15) * LDA + kb + laneHi * 8;
                const v8h lo = *(const v8h*)(p);
                const v8h hi = *(const v8h*)(p + 16);
                afrag[mi] = __builtin_shufflevector(lo, hi,
                    0,1,2,3,4,5,6,7,8,9,10,11,12,13,14,15);
            }

            // B frags: lanes 0-15: N=lane, halves = K(kb+0..7 | kb+16..23); lanes16-31: K(kb+8..15 | kb+24..31)
            #pragma unroll
            for (int ni = 0; ni < 4; ++ni) {
                const _Float16* q = ldsB + (waveN + ni * 16 + lane15) * LDB + kb + laneHi * 8;
                const v8h blo = *(const v8h*)(q);
                const v8h bhi = *(const v8h*)(q + 16);
                const v16h bfrag = __builtin_shufflevector(blo, bhi,
                    0,1,2,3,4,5,6,7,8,9,10,11,12,13,14,15);

                #pragma unroll
                for (int mi = 0; mi < 4; ++mi) {
                    acc[mi][ni] = __builtin_amdgcn_wmma_f32_16x16x32_f16(
                        false, afrag[mi], false, bfrag,
                        (short)0, acc[mi][ni], false, false);
                }
            }
        }

        __syncthreads();
    }

    #pragma unroll
    for (int mi = 0; mi < 4; ++mi) {
        #pragma unroll
        for (int ni = 0; ni < 4; ++ni) {
            #pragma unroll
            for (int r = 0; r < 8; ++r) {
                const int row = mBlock + waveM + mi * 16 + laneHi * 8 + r;
                const int col = nBlock + waveN + ni * 16 + lane15;
                C[(size_t)row * N_DIM + col] = acc[mi][ni][r];
            }
        }
    }
}

// ============================================================================
// Fallback: fused kernel (inline dequant) if workspace is too small.
// Same 128x256 block shape.
// ============================================================================
__global__ __launch_bounds__(THREADS)
void gemm_fused_kernel(const float* __restrict__ A, const int8_t* __restrict__ W,
                       const float* __restrict__ S, float* __restrict__ C)
{
    __shared__ _Float16 ldsA[BM * LDA];
    __shared__ _Float16 ldsB[BN * LDB];

    const int tid    = threadIdx.x;
    const int wid    = tid >> 5;
    const int lane   = tid & 31;
    const int laneHi = lane >> 4;
    const int lane15 = lane & 15;

    const int mBlock = blockIdx.y * BM;
    const int nBlock = blockIdx.x * BN;
    const int waveM  = (wid & 1) * 64;
    const int waveN  = (wid >> 1) * 64;

    v8f acc[4][4] = {};

    // A: 128x64 fp32 -> 2048 float4 / 256 thr = 8 each
    // B: 256x64 int8 -> 1024 x 16B / 256 thr = 4 each
    float4 aReg[8];
    int4   bReg[4];
    float  sReg[4];

    auto loadChunk = [&](int k0) {
        #pragma unroll
        for (int j = 0; j < 8; ++j) {
            const int idx = tid + j * THREADS;
            const int row = idx >> 4;               // 16 float4 per row
            const int c4  = (idx & 15) << 2;
            aReg[j] = *(const float4*)(A + (size_t)(mBlock + row) * K_DIM + k0 + c4);
        }
        const int g = k0 >> 7;
        #pragma unroll
        for (int j = 0; j < 4; ++j) {
            const int idx = tid + j * THREADS;
            const int row = idx >> 2;               // 4 x 16B per row
            const int cc  = (idx & 3) << 4;
            bReg[j] = *(const int4*)(W + (size_t)(nBlock + row) * K_DIM + k0 + cc);
            sReg[j] = S[(size_t)(nBlock + row) * NG + g];
        }
    };

    auto storeChunk = [&]() {
        #pragma unroll
        for (int j = 0; j < 8; ++j) {
            const int idx = tid + j * THREADS;
            const int row = idx >> 4;
            const int c4  = (idx & 15) << 2;
            v4h h;
            h[0] = (_Float16)aReg[j].x; h[1] = (_Float16)aReg[j].y;
            h[2] = (_Float16)aReg[j].z; h[3] = (_Float16)aReg[j].w;
            *(v4h*)(ldsA + row * LDA + c4) = h;
        }
        #pragma unroll
        for (int j = 0; j < 4; ++j) {
            const int idx = tid + j * THREADS;
            const int row = idx >> 2;
            const int cc  = (idx & 3) << 4;
            const int8_t* b = (const int8_t*)&bReg[j];
            const float s = sReg[j];
            v8h h0, h1;
            #pragma unroll
            for (int e = 0; e < 8; ++e) {
                h0[e] = (_Float16)(s * (float)b[e]);
                h1[e] = (_Float16)(s * (float)b[e + 8]);
            }
            _Float16* d = ldsB + row * LDB + cc;
            *(v8h*)(d)     = h0;
            *(v8h*)(d + 8) = h1;
        }
    };

    const int numChunks = K_DIM / BK;
    loadChunk(0);
    for (int kc = 0; kc < numChunks; ++kc) {
        storeChunk();
        __syncthreads();
        if (kc + 1 < numChunks) loadChunk((kc + 1) * BK);

        #pragma unroll
        for (int kk = 0; kk < 2; ++kk) {
            const int kb = kk * 32;
            v16h afrag[4];
            #pragma unroll
            for (int mi = 0; mi < 4; ++mi) {
                const _Float16* p = ldsA + (waveM + mi * 16 + lane15) * LDA + kb + laneHi * 8;
                const v8h lo = *(const v8h*)(p);
                const v8h hi = *(const v8h*)(p + 16);
                afrag[mi] = __builtin_shufflevector(lo, hi,
                    0,1,2,3,4,5,6,7,8,9,10,11,12,13,14,15);
            }
            #pragma unroll
            for (int ni = 0; ni < 4; ++ni) {
                const _Float16* q = ldsB + (waveN + ni * 16 + lane15) * LDB + kb + laneHi * 8;
                const v8h blo = *(const v8h*)(q);
                const v8h bhi = *(const v8h*)(q + 16);
                const v16h bfrag = __builtin_shufflevector(blo, bhi,
                    0,1,2,3,4,5,6,7,8,9,10,11,12,13,14,15);
                #pragma unroll
                for (int mi = 0; mi < 4; ++mi) {
                    acc[mi][ni] = __builtin_amdgcn_wmma_f32_16x16x32_f16(
                        false, afrag[mi], false, bfrag,
                        (short)0, acc[mi][ni], false, false);
                }
            }
        }
        __syncthreads();
    }

    #pragma unroll
    for (int mi = 0; mi < 4; ++mi)
        #pragma unroll
        for (int ni = 0; ni < 4; ++ni)
            #pragma unroll
            for (int r = 0; r < 8; ++r) {
                const int row = mBlock + waveM + mi * 16 + laneHi * 8 + r;
                const int col = nBlock + waveN + ni * 16 + lane15;
                C[(size_t)row * N_DIM + col] = acc[mi][ni][r];
            }
}

// ============================================================================
extern "C" void kernel_launch(void* const* d_in, const int* in_sizes, int n_in,
                              void* d_out, int out_size, void* d_ws, size_t ws_size,
                              hipStream_t stream) {
    const float*  A = (const float*)d_in[0];    // [4096, 4096] fp32
    const int8_t* W = (const int8_t*)d_in[1];   // [11008, 4096] int8
    const float*  S = (const float*)d_in[2];    // [11008, 32] fp32
    float* out = (float*)d_out;                 // [4096, 11008] fp32

    const int M = in_sizes[0] / K_DIM;          // 4096
    dim3 grid(N_DIM / BN, M / BM);              // (43, 32)

    const size_t needA = (size_t)M * K_DIM * sizeof(_Float16);      // 32 MB
    const size_t needW = (size_t)N_DIM * K_DIM * sizeof(_Float16);  // 90 MB

    if (ws_size >= needA + needW) {
        _Float16* Ah = (_Float16*)d_ws;
        _Float16* Wh = Ah + (size_t)M * K_DIM;

        const int total4  = (M * K_DIM) / 4;            // float4 chunks
        const int total16 = (N_DIM * K_DIM) / 16;       // 16B chunks
        convertA_kernel<<<4096, 256, 0, stream>>>(A, Ah, total4);
        dequantW_kernel<<<4096, 256, 0, stream>>>(W, S, Wh, total16);
        gemm_f16_kernel<<<grid, THREADS, 0, stream>>>(Ah, Wh, out);
    } else {
        gemm_fused_kernel<<<grid, THREADS, 0, stream>>>(A, W, S, out);
    }
}
